// _GraphConv_2310692405874
// MI455X (gfx1250) — matrix operand order repaired and checked by reading, not compiled, not run on hardware
//
#include <hip/hip_runtime.h>
#include <math.h>

typedef __attribute__((ext_vector_type(16))) _Float16 v16h;
typedef __attribute__((ext_vector_type(8)))  float    v8f;

#define JNT   17
#define DIN   64
#define DOUT  64
#define NBAT  4               // batches per workgroup
#define RROWS (NBAT * JNT)    // 68 real rows
#define PROWS 80              // padded to 5 tiles of 16
#define NEGV  (-9e15f)

// ---------------------------------------------------------------------------
// Kernel 1: masked adjacency softmax -> diag[17], off[17*17] in workspace
// ---------------------------------------------------------------------------
__global__ __launch_bounds__(32)
void k_adj(const float* __restrict__ e, const int* __restrict__ rows,
           const int* __restrict__ cols, int ne, float* __restrict__ ws) {
  __shared__ float adj[JNT * JNT];
  int t = threadIdx.x;
  for (int i = t; i < JNT * JNT; i += 32) adj[i] = NEGV;
  __syncthreads();
  for (int i = t; i < ne; i += 32) adj[rows[i] * JNT + cols[i]] = e[i];
  __syncthreads();
  if (t < JNT) {
    float mx = -3.4e38f;
    for (int j = 0; j < JNT; ++j) mx = fmaxf(mx, adj[t * JNT + j]);
    float sum = 0.f;
    for (int j = 0; j < JNT; ++j) sum += expf(adj[t * JNT + j] - mx);
    float inv = 1.f / sum;
    ws[t] = expf(adj[t * JNT + t] - mx) * inv;                    // diag
    for (int j = 0; j < JNT; ++j)
      ws[32 + t * JNT + j] = (j == t) ? 0.f : expf(adj[t * JNT + j] - mx) * inv;
  }
}

// ---------------------------------------------------------------------------
// Kernel 2: per 4-batch group: H = x @ [W0|W1] via WMMA (f16 in, f32 acc),
// then fused adjacency mix + bias, write pre-BN to d_out, and emit
// deterministic per-block channel sums / sums-of-squares.
// ---------------------------------------------------------------------------
__global__ __launch_bounds__(256)
void k_gemm_mix(const float* __restrict__ x, const float* __restrict__ Wg,
                const float* __restrict__ bias, const float* __restrict__ wsadj,
                float* __restrict__ pre, float* __restrict__ P,
                float* __restrict__ Q) {
  __shared__ _Float16 Ax[PROWS][DIN];       // 10240 B
  __shared__ float    H[PROWS][2 * DOUT];   // 40960 B  [h0 | h1]
  __shared__ float    dg[JNT];
  __shared__ float    offm[JNT * JNT];
  __shared__ float    red[2][NBAT][DOUT];   // 2048 B

  const int tid = threadIdx.x;
  const int g   = blockIdx.x;

  // Stage x (f32 -> f16) into LDS, zero-pad rows 68..79
  for (int idx = tid; idx < PROWS * DIN; idx += 256) {
    int r = idx >> 6;
    float v = (r < RROWS) ? x[(size_t)g * (RROWS * DIN) + idx] : 0.f;
    Ax[r][idx & 63] = (_Float16)v;
  }
  for (int idx = tid; idx < JNT; idx += 256)       dg[idx]   = wsadj[idx];
  for (int idx = tid; idx < JNT * JNT; idx += 256) offm[idx] = wsadj[32 + idx];
  __syncthreads();

  const int wave = tid >> 5;          // one wave per 16-col tile of N=128
  const int lane = tid & 31;
  const int lh   = lane & 15;
  const int hi   = lane >> 4;
  const int n0   = wave * 16;
  // columns 0..63 -> W[0], 64..127 -> W[1]; each is 64x64 row-major
  const float* Wn = Wg + ((n0 >= 64) ? (64 * 64 + (n0 - 64)) : n0);

  // B fragments (K x N = 32x16 each): N = n0+lh; lanes 0-15 hold
  // K={0..7,16..23}, lanes 16-31 hold K={8..15,24..31}; VGPR v packs K=pair.
  v16h bf0, bf1;
  {
#pragma unroll
    for (int v = 0; v < 8; ++v) {
      int kb = ((v < 4) ? (2 * v) : (16 + 2 * (v - 4))) + (hi ? 8 : 0);
      bf0[2 * v]     = (_Float16)Wn[kb * DOUT + lh];
      bf0[2 * v + 1] = (_Float16)Wn[(kb + 1) * DOUT + lh];
      bf1[2 * v]     = (_Float16)Wn[(32 + kb) * DOUT + lh];
      bf1[2 * v + 1] = (_Float16)Wn[(32 + kb + 1) * DOUT + lh];
    }
  }

  for (int rt = 0; rt < PROWS / 16; ++rt) {
    const int m = rt * 16 + lh;
    // A fragment (16x32 f16): lanes 0-15: M=lane, K={0..7,16..23};
    // lanes 16-31: M=lane-16, K={8..15,24..31}; VGPR g packs K=pair.
    v16h a0, a1;
#pragma unroll
    for (int gg = 0; gg < 8; ++gg) {
      int kb = ((gg < 4) ? (2 * gg) : (16 + 2 * (gg - 4))) + (hi ? 8 : 0);
      a0[2 * gg]     = Ax[m][kb];
      a0[2 * gg + 1] = Ax[m][kb + 1];
      a1[2 * gg]     = Ax[m][32 + kb];
      a1[2 * gg + 1] = Ax[m][32 + kb + 1];
    }
    v8f c = {};
    c = __builtin_amdgcn_wmma_f32_16x16x32_f16(false, a0, false, bf0,
                                               (short)0, c, false, false);
    c = __builtin_amdgcn_wmma_f32_16x16x32_f16(false, a1, false, bf1,
                                               (short)0, c, false, false);
    // C/D layout: VGPR v -> M = v + (hi?8:0), N = lh
#pragma unroll
    for (int v = 0; v < 8; ++v)
      H[rt * 16 + v + (hi ? 8 : 0)][n0 + lh] = c[v];
  }
  __syncthreads();

  // Fused mix + bias; thread = (batch q, channel f); deterministic sums.
  const int f = tid & 63;
  const int q = tid >> 6;   // NBAT==4 quarters == batch-in-group
  const float bs = bias[f];
  float s = 0.f, s2 = 0.f;
  for (int i = 0; i < JNT; ++i) {
    float acc = dg[i] * H[q * JNT + i][f] + bs;
    for (int j = 0; j < JNT; ++j)
      acc += offm[i * JNT + j] * H[q * JNT + j][DOUT + f];
    pre[(size_t)g * (RROWS * DOUT) + (q * JNT + i) * DOUT + f] = acc;
    s  += acc;
    s2 += acc * acc;
  }
  red[0][q][f] = s;
  red[1][q][f] = s2;
  __syncthreads();
  if (q == 0) {
    float ts = red[0][0][f] + red[0][1][f] + red[0][2][f] + red[0][3][f];
    float t2 = red[1][0][f] + red[1][1][f] + red[1][2][f] + red[1][3][f];
    P[(size_t)g * DOUT + f] = ts;
    Q[(size_t)g * DOUT + f] = t2;
  }
}

// ---------------------------------------------------------------------------
// Kernel 3: deterministic reduction of per-block partials -> scale/shift
// ---------------------------------------------------------------------------
__global__ __launch_bounds__(256)
void k_stats(const float* __restrict__ P, const float* __restrict__ Q,
             const float* __restrict__ gamma, const float* __restrict__ beta,
             int nblk, float ninv, float* __restrict__ sc,
             float* __restrict__ sh) {
  __shared__ float red[2][4][DOUT];
  const int f = threadIdx.x & 63;
  const int q = threadIdx.x >> 6;
  float s = 0.f, s2 = 0.f;
  for (int b = q; b < nblk; b += 4) {
    s  += P[(size_t)b * DOUT + f];
    s2 += Q[(size_t)b * DOUT + f];
  }
  red[0][q][f] = s;
  red[1][q][f] = s2;
  __syncthreads();
  if (q == 0) {
    float S  = red[0][0][f] + red[0][1][f] + red[0][2][f] + red[0][3][f];
    float S2 = red[1][0][f] + red[1][1][f] + red[1][2][f] + red[1][3][f];
    float mean = S * ninv;
    float var  = S2 * ninv - mean * mean;
    float a = gamma[f] * rsqrtf(var + 1e-5f);
    sc[f] = a;
    sh[f] = beta[f] - mean * a;
  }
}

// ---------------------------------------------------------------------------
// Kernel 4: in-place normalize + ReLU over d_out (float4 vectorized)
// ---------------------------------------------------------------------------
__global__ __launch_bounds__(256)
void k_norm(float* __restrict__ out, const float* __restrict__ sc,
            const float* __restrict__ sh, int n4) {
  int idx = blockIdx.x * 256 + threadIdx.x;
  if (idx >= n4) return;
  float4 v = reinterpret_cast<float4*>(out)[idx];
  float4 a = reinterpret_cast<const float4*>(sc)[idx & 15];
  float4 c = reinterpret_cast<const float4*>(sh)[idx & 15];
  v.x = fmaxf(v.x * a.x + c.x, 0.f);
  v.y = fmaxf(v.y * a.y + c.y, 0.f);
  v.z = fmaxf(v.z * a.z + c.z, 0.f);
  v.w = fmaxf(v.w * a.w + c.w, 0.f);
  reinterpret_cast<float4*>(out)[idx] = v;
}

// ---------------------------------------------------------------------------
extern "C" void kernel_launch(void* const* d_in, const int* in_sizes, int n_in,
                              void* d_out, int out_size, void* d_ws,
                              size_t ws_size, hipStream_t stream) {
  const float* x     = (const float*)d_in[0];
  const float* W     = (const float*)d_in[1];
  const float* e     = (const float*)d_in[2];
  const float* bias  = (const float*)d_in[3];
  const float* gamma = (const float*)d_in[4];
  const float* beta  = (const float*)d_in[5];
  const int*   rows  = (const int*)d_in[6];
  const int*   cols  = (const int*)d_in[7];
  const int ne = in_sizes[2];

  const int Bn   = in_sizes[0] / (JNT * DIN);   // 16384
  const int ngrp = Bn / NBAT;                   // 4096 workgroups

  float* ws = (float*)d_ws;
  // ws layout (floats): [0..16] diag, [32..320] off, [384..447] scale,
  // [448..511] shift, [512..) per-block partial sums then sumsqs.
  float* sc = ws + 384;
  float* sh = ws + 448;
  float* P  = ws + 512;
  float* Q  = P + (size_t)ngrp * DOUT;

  k_adj<<<1, 32, 0, stream>>>(e, rows, cols, ne, ws);
  k_gemm_mix<<<ngrp, 256, 0, stream>>>(x, W, bias, ws, (float*)d_out, P, Q);
  const float ninv = 1.0f / (float)(Bn * JNT);
  k_stats<<<1, 256, 0, stream>>>(P, Q, gamma, beta, ngrp, ninv, sc, sh);
  const int n4 = (Bn * JNT * DOUT) / 4;
  k_norm<<<(n4 + 255) / 256, 256, 0, stream>>>((float*)d_out, sc, sh, n4);
}
